// FSDPCompatibleMultiheadAttention_41171556499594
// MI455X (gfx1250) — matrix operand order repaired and checked by reading, not compiled, not run on hardware
//
#include <hip/hip_runtime.h>

// ---------------------------------------------------------------------------
// CDNA5 (gfx1250) multi-head attention forward, bf16 WMMA everywhere.
// B=2, S=2048, D=1024, H=16, Hd=64.  wave32.
// ---------------------------------------------------------------------------

typedef __attribute__((ext_vector_type(16))) __bf16 bf16x16;
typedef __attribute__((ext_vector_type(8)))  float  f32x8;
typedef int v4i_16b __attribute__((vector_size(16)));   // b128 payload type

union BfTile {                 // one WMMA operand: 32 bytes per lane
  bf16x16 v;
  uint4   u[2];
};

// Async global->LDS copy (CDNA5 GLOBAL_LOAD_ASYNC_TO_LDS_B128, ASYNCcnt).
#if defined(__gfx1250__) && __has_builtin(__builtin_amdgcn_global_load_async_to_lds_b128)
#define ASYNC_LDS 1
#else
#define ASYNC_LDS 0
#endif

__device__ __forceinline__ void async_ld16(const unsigned short* g,
                                           unsigned short* l) {
#if ASYNC_LDS
  __builtin_amdgcn_global_load_async_to_lds_b128(
      (__attribute__((address_space(1))) v4i_16b*)(g),
      (__attribute__((address_space(3))) v4i_16b*)(l), 0, 0);
#endif
}
__device__ __forceinline__ void wait_async0() {
#if ASYNC_LDS
#if __has_builtin(__builtin_amdgcn_s_wait_asynccnt)
  __builtin_amdgcn_s_wait_asynccnt(0);
#else
  asm volatile("s_wait_asynccnt 0x0" ::: "memory");
#endif
#endif
}

__device__ __forceinline__ unsigned short f32_to_bf16(float f) {
  unsigned int u = __float_as_uint(f);
  u += 0x7FFFu + ((u >> 16) & 1u);          // round-to-nearest-even
  return (unsigned short)(u >> 16);
}
__device__ __forceinline__ float bf16_to_f32(unsigned short h) {
  return __uint_as_float(((unsigned int)h) << 16);
}
__device__ __forceinline__ f32x8 zero8() {
  f32x8 z;
#pragma unroll
  for (int i = 0; i < 8; ++i) z[i] = 0.0f;
  return z;
}

// ---------------------------------------------------------------------------
// Kernel 0: f32 -> bf16 conversion, 4 elements/thread (b128 in, b64 out)
// ---------------------------------------------------------------------------
__global__ void cvt_f32_to_bf16_x4(const float* __restrict__ in,
                                   unsigned short* __restrict__ out, int n4) {
  int i = blockIdx.x * blockDim.x + threadIdx.x;
  if (i < n4) {
    float4 f = ((const float4*)in)[i];
    ushort4 o;
    o.x = f32_to_bf16(f.x);
    o.y = f32_to_bf16(f.y);
    o.z = f32_to_bf16(f.z);
    o.w = f32_to_bf16(f.w);
    ((ushort4*)out)[i] = o;
  }
}

// ---------------------------------------------------------------------------
// Kernel 1: GEMM  Y[m,n] = sum_k A[m,k] * W[n,k] + bias[n]
//   A: bf16 [4096,1024] row-major, W: bf16 [1024,1024] row-major (N,K)
//   Block: 256 threads = 8 waves sharing one 64-wide N tile.
//   W k-chunk (64x32 bf16 = 4KB) staged into a DOUBLE-BUFFERED LDS panel
//   via GLOBAL_LOAD_ASYNC_TO_LDS_B128; the copy of chunk k+1 overlaps the
//   4 WMMAs of chunk k; one barrier per chunk.
//   mode 0: store bf16 [B,H,S,Hd]   (Q, K)
//   mode 1: store bf16 [B,H,Hd,S]   (V, transposed per head)
//   mode 2: store f32  [M,1024]     (final output projection)
// ---------------------------------------------------------------------------
__global__ __launch_bounds__(256) void gemm_bf16_wmma(
    const unsigned short* __restrict__ A,
    const unsigned short* __restrict__ W,
    const float* __restrict__ bias,
    unsigned short* __restrict__ out_bf,
    float* __restrict__ out_f32,
    int mode) {
  __shared__ unsigned short w_s[2][64 * 32];   // 2 x 4KB W chunk

  const int tid  = threadIdx.x;
  const int lane = tid & 31;
  const int wv   = tid >> 5;
  const int hi   = lane >> 4;    // which 16-lane half
  const int ln   = lane & 15;

  const int nt = blockIdx.x >> 5;                 // 0..15  (N tile of 64)
  const int mt = ((blockIdx.x & 31) << 3) + wv;   // 0..255 (M tile of 16)

  f32x8 acc[4];
#pragma unroll
  for (int j = 0; j < 4; ++j) acc[j] = zero8();

  // cooperative W staging mapping: one b128 per thread per chunk
  const int crow = tid >> 2, cpart = tid & 3;
  const unsigned short* wsrc =
      W + (size_t)(nt * 64 + crow) * 1024 + cpart * 8;
  unsigned short* ldst0 = &w_s[0][crow * 32 + cpart * 8];
  unsigned short* ldst1 = &w_s[1][crow * 32 + cpart * 8];

#if ASYNC_LDS
  async_ld16(wsrc, ldst0);     // prefetch first chunk
#endif

  const unsigned short* arowbase = A + (size_t)(mt * 16 + ln) * 1024 + hi * 8;

  for (int kc = 0; kc < 1024; kc += 32) {
    const int cur = (kc >> 5) & 1;
#if ASYNC_LDS
    wait_async0();             // own wave's async writes done
    __syncthreads();           // everyone's writes done, prev readers done
    if (kc + 32 < 1024)
      async_ld16(wsrc + kc + 32, cur ? ldst0 : ldst1);
    const unsigned short* wbuf = w_s[cur];
#else
    __syncthreads();
    *(uint4*)ldst0 = *(const uint4*)(wsrc + kc);
    __syncthreads();
    const unsigned short* wbuf = w_s[0];
#endif

    // A operand: lane ln holds row M=ln; K = {kc+hi*8..+7, kc+16+hi*8..+7}
    BfTile a;
    const unsigned short* arow = arowbase + kc;
    a.u[0] = *(const uint4*)(arow);
    a.u[1] = *(const uint4*)(arow + 16);

#pragma unroll
    for (int j = 0; j < 4; ++j) {
      // B operand: lane ln = column n; K = {hi*8..+7, 16+hi*8..+7}
      BfTile b;
      const unsigned short* bp = &wbuf[(j * 16 + ln) * 32 + hi * 8];
      b.u[0] = *(const uint4*)(bp);
      b.u[1] = *(const uint4*)(bp + 16);
      acc[j] = __builtin_amdgcn_wmma_f32_16x16x32_bf16(
          false, a.v, false, b.v, (short)0, acc[j], false, false);
    }
  }

  // Epilogue: C elem r -> (M = mt*16 + r + hi*8, N = nt*64 + j*16 + ln)
#pragma unroll
  for (int j = 0; j < 4; ++j) {
    const int n  = nt * 64 + j * 16 + ln;
    const float bj = bias[n];
    const int h = n >> 6, hd = n & 63;
#pragma unroll
    for (int r = 0; r < 8; ++r) {
      const int m = mt * 16 + r + hi * 8;
      const float val = acc[j][r] + bj;
      if (mode == 2) {
        out_f32[(size_t)m * 1024 + n] = val;
      } else {
        const int bb = m >> 11, s = m & 2047;
        const size_t addr =
            (mode == 0)
                ? ((size_t)((bb * 16 + h) * 2048 + s)) * 64 + hd   // [B,H,S,Hd]
                : ((size_t)((bb * 16 + h) * 64 + hd)) * 2048 + s;  // [B,H,Hd,S]
        out_bf[addr] = f32_to_bf16(val);
      }
    }
  }
}

// ---------------------------------------------------------------------------
// Kernel 2: attention for one (b,h): softmax(Q K^T / 8) V
//   2 waves / block, each wave owns 16 query rows and a private 16x2048
//   bf16 score panel in LDS (64KB/wave; 128KB dynamic LDS per block,
//   within the 320KB WGP budget).
// ---------------------------------------------------------------------------
__global__ __launch_bounds__(64) void attn_bf16_wmma(
    const unsigned short* __restrict__ Qp,   // [B,H,S,64]
    const unsigned short* __restrict__ Kp,   // [B,H,S,64]
    const unsigned short* __restrict__ Vt,   // [B,H,64,S]
    unsigned short* __restrict__ Ctx) {      // [B,S,H,64]
  extern __shared__ unsigned short smem[];
  const int S = 2048;

  const int lane = threadIdx.x & 31;
  const int wv   = threadIdx.x >> 5;
  const int hi   = lane >> 4, ln = lane & 15;

  const int bh = blockIdx.x >> 6;     // 0..31  (b*16 + h)
  const int qb = blockIdx.x & 63;     // 0..63
  const int q0 = qb * 32 + wv * 16;   // first query row of this wave

  unsigned short* sc = smem + wv * 16 * S;            // 16 x 2048 bf16 panel
  float* scratch = (float*)(smem + 2 * 16 * S);
  float* smax = scratch + wv * 32;
  float* sinv = smax + 16;

  const size_t base = (size_t)bh * S * 64;

  // Q operand tiles (2 k-chunks of 32 over Hd=64), kept in registers
  BfTile qa[2];
  {
    const unsigned short* qrow = Qp + base + (size_t)(q0 + ln) * 64;
#pragma unroll
    for (int c = 0; c < 2; ++c) {
      qa[c].u[0] = *(const uint4*)(qrow + c * 32 + hi * 8);
      qa[c].u[1] = *(const uint4*)(qrow + c * 32 + 16 + hi * 8);
    }
  }

  float rmax[8];
#pragma unroll
  for (int r = 0; r < 8; ++r) rmax[r] = -3.0e38f;
  const float scale = 0.125f;   // 1/sqrt(64)

  // ---- Phase 1: scores -> LDS (bf16), running per-row max ----
  for (int kb = 0; kb < S; kb += 16) {
    f32x8 c = zero8();
#pragma unroll
    for (int ch = 0; ch < 2; ++ch) {
      BfTile b;   // B: lane ln = key row kb+ln; K = d, contiguous in Kp
      const unsigned short* krow =
          Kp + base + (size_t)(kb + ln) * 64 + ch * 32 + hi * 8;
      b.u[0] = *(const uint4*)(krow);
      b.u[1] = *(const uint4*)(krow + 16);
      c = __builtin_amdgcn_wmma_f32_16x16x32_bf16(
          false, qa[ch].v, false, b.v, (short)0, c, false, false);
    }
#pragma unroll
    for (int r = 0; r < 8; ++r) {
      const float s = c[r] * scale;
      rmax[r] = fmaxf(rmax[r], s);
      sc[(r + hi * 8) * S + kb + ln] = f32_to_bf16(s);
    }
  }

  // reduce row max across the 16-lane half (rows stay separate per half)
#pragma unroll
  for (int r = 0; r < 8; ++r) {
#pragma unroll
    for (int m = 1; m < 16; m <<= 1)
      rmax[r] = fmaxf(rmax[r], __shfl_xor(rmax[r], m, 32));
  }
  if (lane == 0) {
#pragma unroll
    for (int r = 0; r < 8; ++r) smax[r] = rmax[r];
  }
  if (lane == 16) {
#pragma unroll
    for (int r = 0; r < 8; ++r) smax[8 + r] = rmax[r];
  }

  // ---- Phase 2: exp2 numerators in LDS + row sums (b64 vectorized) ----
  {
    const int row = ln;                 // 2 lanes per row, 1024 cols each
    const float rm = smax[row];
    const float l2e = 1.44269504f;
    float sum = 0.0f;
    uint2* p = (uint2*)(sc + row * S + hi * 1024);
    for (int i = 0; i < 256; ++i) {     // 4 bf16 per iteration
      uint2 u = p[i];
      float e0 = exp2f((bf16_to_f32((unsigned short)(u.x & 0xFFFF)) - rm) * l2e);
      float e1 = exp2f((bf16_to_f32((unsigned short)(u.x >> 16)) - rm) * l2e);
      float e2 = exp2f((bf16_to_f32((unsigned short)(u.y & 0xFFFF)) - rm) * l2e);
      float e3 = exp2f((bf16_to_f32((unsigned short)(u.y >> 16)) - rm) * l2e);
      sum += (e0 + e1) + (e2 + e3);
      u.x = (unsigned)f32_to_bf16(e0) | ((unsigned)f32_to_bf16(e1) << 16);
      u.y = (unsigned)f32_to_bf16(e2) | ((unsigned)f32_to_bf16(e3) << 16);
      p[i] = u;
    }
    sum += __shfl_xor(sum, 16, 32);
    if (lane < 16) sinv[row] = 1.0f / sum;
  }

  // ---- Phase 3: P x V ----
  f32x8 acc[4];
#pragma unroll
  for (int j = 0; j < 4; ++j) acc[j] = zero8();

  for (int kc = 0; kc < S; kc += 32) {
    BfTile a;   // A: lane ln = query row; K = keys, read from LDS panel
    const unsigned short* prow = sc + ln * S + kc + hi * 8;
    a.u[0] = *(const uint4*)(prow);
    a.u[1] = *(const uint4*)(prow + 16);
#pragma unroll
    for (int j = 0; j < 4; ++j) {
      BfTile b;   // B: lane ln = output col hd; V^T rows are key-contiguous
      const unsigned short* vrow =
          Vt + base + (size_t)(j * 16 + ln) * S + kc + hi * 8;
      b.u[0] = *(const uint4*)(vrow);
      b.u[1] = *(const uint4*)(vrow + 16);
      acc[j] = __builtin_amdgcn_wmma_f32_16x16x32_bf16(
          false, a.v, false, b.v, (short)0, acc[j], false, false);
    }
  }

  // Epilogue: normalize rows, store Ctx[B,S,H,Hd] as bf16
  const int bb = bh >> 4, h = bh & 15;
  float inv[8];
#pragma unroll
  for (int r = 0; r < 8; ++r) inv[r] = sinv[r + hi * 8];
#pragma unroll
  for (int j = 0; j < 4; ++j) {
    const int hd = j * 16 + ln;
#pragma unroll
    for (int r = 0; r < 8; ++r) {
      const int q = q0 + r + hi * 8;
      Ctx[((size_t)(bb * 2048 + q) * 16 + h) * 64 + hd] =
          f32_to_bf16(acc[j][r] * inv[r]);
    }
  }
}

// ---------------------------------------------------------------------------
// Host-side orchestration
// ---------------------------------------------------------------------------
extern "C" void kernel_launch(void* const* d_in, const int* in_sizes, int n_in,
                              void* d_out, int out_size, void* d_ws,
                              size_t ws_size, hipStream_t stream) {
  const float* q  = (const float*)d_in[0];
  const float* k  = (const float*)d_in[1];
  const float* v  = (const float*)d_in[2];
  const float* Wq = (const float*)d_in[3];
  const float* bq = (const float*)d_in[4];
  const float* Wk = (const float*)d_in[5];
  const float* bk = (const float*)d_in[6];
  const float* Wv = (const float*)d_in[7];
  const float* bv = (const float*)d_in[8];
  const float* Wo = (const float*)d_in[9];
  const float* bo = (const float*)d_in[10];
  float* out = (float*)d_out;

  char* ws = (char*)d_ws;
  const size_t MB = 1024 * 1024;
  unsigned short* Xq  = (unsigned short*)(ws + 0 * MB);   // 8MB each
  unsigned short* Xk  = (unsigned short*)(ws + 8 * MB);
  unsigned short* Xv  = (unsigned short*)(ws + 16 * MB);
  unsigned short* Wqb = (unsigned short*)(ws + 24 * MB);  // 2MB each
  unsigned short* Wkb = (unsigned short*)(ws + 26 * MB);
  unsigned short* Wvb = (unsigned short*)(ws + 28 * MB);
  unsigned short* Wob = (unsigned short*)(ws + 30 * MB);
  unsigned short* Qp  = (unsigned short*)(ws + 32 * MB);  // [B,H,S,Hd]
  unsigned short* Kp  = (unsigned short*)(ws + 40 * MB);  // [B,H,S,Hd]
  unsigned short* Vt  = (unsigned short*)(ws + 48 * MB);  // [B,H,Hd,S]
  unsigned short* Ctx = (unsigned short*)(ws + 56 * MB);  // [B,S,H,64]

  const int NX4 = 4096 * 1024 / 4;   // activations, 4 elems/thread
  const int NW4 = 1024 * 1024 / 4;   // weights

  cvt_f32_to_bf16_x4<<<(NX4 + 255) / 256, 256, 0, stream>>>(q, Xq, NX4);
  cvt_f32_to_bf16_x4<<<(NX4 + 255) / 256, 256, 0, stream>>>(k, Xk, NX4);
  cvt_f32_to_bf16_x4<<<(NX4 + 255) / 256, 256, 0, stream>>>(v, Xv, NX4);
  cvt_f32_to_bf16_x4<<<(NW4 + 255) / 256, 256, 0, stream>>>(Wq, Wqb, NW4);
  cvt_f32_to_bf16_x4<<<(NW4 + 255) / 256, 256, 0, stream>>>(Wk, Wkb, NW4);
  cvt_f32_to_bf16_x4<<<(NW4 + 255) / 256, 256, 0, stream>>>(Wv, Wvb, NW4);
  cvt_f32_to_bf16_x4<<<(NW4 + 255) / 256, 256, 0, stream>>>(Wo, Wob, NW4);

  // QKV projections: 512 blocks x 256 threads (8 waves sharing one N tile)
  gemm_bf16_wmma<<<512, 256, 0, stream>>>(Xq, Wqb, bq, Qp, nullptr, 0);
  gemm_bf16_wmma<<<512, 256, 0, stream>>>(Xk, Wkb, bk, Kp, nullptr, 0);
  gemm_bf16_wmma<<<512, 256, 0, stream>>>(Xv, Wvb, bv, Vt, nullptr, 1);

  // Attention: 2048 blocks x 64 threads, 128KB+256B dynamic LDS
  const size_t smem = 2 * 16 * 2048 * sizeof(unsigned short) +
                      2 * 32 * sizeof(float);
  attn_bf16_wmma<<<2048, 64, smem, stream>>>(Qp, Kp, Vt, Ctx);

  // Output projection -> f32 d_out
  gemm_bf16_wmma<<<512, 256, 0, stream>>>(Ctx, Wob, bo, nullptr, out, 2);
}
